// ForwardModel_78443282694508
// MI455X (gfx1250) — compile-verified, matrix operand order checked by reading, not run on hardware
//
#include <hip/hip_runtime.h>
#include <hip/hip_bf16.h>
#include <math.h>
#include <stdint.h>

// ---------------------------------------------------------------------------
// Problem constants (from the reference)
// ---------------------------------------------------------------------------
#define IN_DIM 19296
#define H      512
#define G      2048      // 4*H
#define OUT_N  11
#define STEPS  1024
#define NWG    16        // workgroups in the recurrence grid

// GEMM tiling
#define BM 64            // rows of C per block
#define BN 128           // cols of C per block
#define BK 32            // k-step (WMMA K)
#define LDSA_STRIDE 36   // fp32 elements; 144B rows: 16B-aligned, bank-skewed

typedef __attribute__((ext_vector_type(16))) __bf16 v16bf;
typedef __attribute__((ext_vector_type(8)))  float  v8f;
typedef int v4i_b128 __attribute__((vector_size(16)));   // b128 transfer unit

// ---------------------------------------------------------------------------
// Optional CDNA5 async memory->LDS staging (ASYNCcnt-tracked, bypasses VGPRs)
// ---------------------------------------------------------------------------
#if defined(__has_builtin)
#if __has_builtin(__builtin_amdgcn_global_load_async_to_lds_b128) && \
    __has_builtin(__builtin_amdgcn_s_wait_asynccnt)
#define USE_ASYNC_LDS 1
#endif
#endif
#ifndef USE_ASYNC_LDS
#define USE_ASYNC_LDS 0
#endif

#if USE_ASYNC_LDS
__device__ __forceinline__ void async_copy_b128(const float* g, float* l) {
  // Builtin signature (probed): (v4i __device__*, v4i __shared__*, imm, imm).
  // Build address-space pointers via integer casts (generic LDS addr low 32
  // bits == LDS byte offset; generic global addr == global addr).
  __builtin_amdgcn_global_load_async_to_lds_b128(
      (__attribute__((address_space(1))) v4i_b128*)(uintptr_t)g,
      (__attribute__((address_space(3))) v4i_b128*)(uint32_t)(uintptr_t)l,
      0, 0);
}
#endif

// ---------------------------------------------------------------------------
// WMMA bf16 fragment loader from an fp32 row (global or LDS).
// CDNA5 16-bit 16x32 A/B layout (ISA 7.12.2):
//   lane L<16 : row=L,    K = {0..7, 16..23}
//   lane L>=16: row=L-16, K = {8..15, 24..31}
// rowbase must be 16B aligned with K contiguous.
// ---------------------------------------------------------------------------
__device__ __forceinline__ v16bf load_frag_f32(const float* rowbase, int half) {
  const float4 a = *reinterpret_cast<const float4*>(rowbase + half * 8 + 0);
  const float4 b = *reinterpret_cast<const float4*>(rowbase + half * 8 + 4);
  const float4 c = *reinterpret_cast<const float4*>(rowbase + 16 + half * 8 + 0);
  const float4 d = *reinterpret_cast<const float4*>(rowbase + 16 + half * 8 + 4);
  v16bf r;
  r[0]  = (__bf16)a.x; r[1]  = (__bf16)a.y; r[2]  = (__bf16)a.z; r[3]  = (__bf16)a.w;
  r[4]  = (__bf16)b.x; r[5]  = (__bf16)b.y; r[6]  = (__bf16)b.z; r[7]  = (__bf16)b.w;
  r[8]  = (__bf16)c.x; r[9]  = (__bf16)c.y; r[10] = (__bf16)c.z; r[11] = (__bf16)c.w;
  r[12] = (__bf16)d.x; r[13] = (__bf16)d.y; r[14] = (__bf16)d.z; r[15] = (__bf16)d.w;
  return r;
}

__device__ __forceinline__ v8f wmma_bf16(v16bf a, v16bf b, v8f c) {
  return __builtin_amdgcn_wmma_f32_16x16x32_bf16(
      /*neg_a=*/false, a, /*neg_b=*/false, b,
      /*c_mod=*/(short)0, c, /*reuse_a=*/false, /*reuse_b=*/false);
}

// ---------------------------------------------------------------------------
// C[M x N] = A[M x K] * W[N x K]^T + bias[N], fp32 in/out, bf16 WMMA compute.
// Block = 256 threads (8 waves), tile BM x BN = 64 x 128, k-step 32.
// Async path: double-buffered LDS; each iteration issues the next tile's
// 6 async b128 copies per wave, then s_wait_asynccnt 6 (in-order completion
// => previous tile landed) overlaps HBM streaming with WMMA compute.
// Each wave owns a 32x32 sub-tile = 2x2 accumulators -> 4 v_wmma per k-step.
// Exact tiling everywhere, EXEC stays all-ones (WMMA requirement).
// ---------------------------------------------------------------------------
__global__ void __launch_bounds__(256)
wmma_gemm_tiled(const float* __restrict__ A,
                const float* __restrict__ W,
                const float* __restrict__ bias,
                float* __restrict__ C,
                int M, int N, int K) {
#if USE_ASYNC_LDS
  __shared__ float As[2][BM * LDSA_STRIDE];   // 2 x  9.2 KB
  __shared__ float Bs[2][BN * LDSA_STRIDE];   // 2 x 18.4 KB
#else
  __shared__ float As[1][BM * LDSA_STRIDE];
  __shared__ float Bs[1][BN * LDSA_STRIDE];
#endif

  const int tid  = threadIdx.x;            // 0..255
  const int wave = tid >> 5;               // 0..7
  const int lane = tid & 31;
  const int half = lane >> 4;
  const int lr   = lane & 15;

  const int blockM = blockIdx.y * BM;
  const int blockN = blockIdx.x * BN;

  const int wm = (wave & 1) * 32;          // wave M offset in tile (0|32)
  const int wn = (wave >> 1) * 32;         // wave N offset in tile (0..96)

  // Cooperative staging assignments (fp32, contiguous 16B chunks)
  const int arow = tid >> 2;               // 0..63
  const int acol = (tid & 3) * 8;          // 0,8,16,24
  const int brow = tid >> 1;               // 0..127
  const int bcol = (tid & 1) * 16;         // 0,16

  const float* __restrict__ Ag = A + (size_t)(blockM + arow) * K + acol;
  const float* __restrict__ Wg = W + (size_t)(blockN + brow) * K + bcol;
  const int aoff = arow * LDSA_STRIDE + acol;
  const int boff = brow * LDSA_STRIDE + bcol;

  v8f acc00 = {}, acc01 = {}, acc10 = {}, acc11 = {};

#if USE_ASYNC_LDS
  auto issue_tile = [&](int buf, int k0) {
    float* const Al = &As[buf][aoff];
    float* const Bl = &Bs[buf][boff];
    async_copy_b128(Ag + k0 + 0,  Al + 0);
    async_copy_b128(Ag + k0 + 4,  Al + 4);
    async_copy_b128(Wg + k0 + 0,  Bl + 0);
    async_copy_b128(Wg + k0 + 4,  Bl + 4);
    async_copy_b128(Wg + k0 + 8,  Bl + 8);
    async_copy_b128(Wg + k0 + 12, Bl + 12);
  };

  issue_tile(0, 0);                        // prologue: 6 ops in flight / wave
  int buf = 0;
  for (int k0 = 0; k0 < K; k0 += BK, buf ^= 1) {
    if (k0 + BK < K) {
      issue_tile(buf ^ 1, k0 + BK);        // 12 in flight
      __builtin_amdgcn_s_wait_asynccnt(6); // current tile's 6 have landed
    } else {
      __builtin_amdgcn_s_wait_asynccnt(0); // epilogue: drain
    }
    __syncthreads();

    const v16bf aF0 = load_frag_f32(&As[buf][(wm + 0  + lr) * LDSA_STRIDE], half);
    const v16bf aF1 = load_frag_f32(&As[buf][(wm + 16 + lr) * LDSA_STRIDE], half);
    const v16bf bF0 = load_frag_f32(&Bs[buf][(wn + 0  + lr) * LDSA_STRIDE], half);
    const v16bf bF1 = load_frag_f32(&Bs[buf][(wn + 16 + lr) * LDSA_STRIDE], half);

    acc00 = wmma_bf16(aF0, bF0, acc00);
    acc01 = wmma_bf16(aF0, bF1, acc01);
    acc10 = wmma_bf16(aF1, bF0, acc10);
    acc11 = wmma_bf16(aF1, bF1, acc11);
    __syncthreads();                       // frees this buffer for re-issue
  }
#else
  float* const Al = &As[0][aoff];
  float* const Bl = &Bs[0][boff];
  for (int k0 = 0; k0 < K; k0 += BK) {
    __builtin_prefetch(Ag + k0 + 256, 0, 1);
    __builtin_prefetch(Wg + k0 + 256, 0, 1);
    *reinterpret_cast<float4*>(Al + 0)  = *reinterpret_cast<const float4*>(Ag + k0 + 0);
    *reinterpret_cast<float4*>(Al + 4)  = *reinterpret_cast<const float4*>(Ag + k0 + 4);
    *reinterpret_cast<float4*>(Bl + 0)  = *reinterpret_cast<const float4*>(Wg + k0 + 0);
    *reinterpret_cast<float4*>(Bl + 4)  = *reinterpret_cast<const float4*>(Wg + k0 + 4);
    *reinterpret_cast<float4*>(Bl + 8)  = *reinterpret_cast<const float4*>(Wg + k0 + 8);
    *reinterpret_cast<float4*>(Bl + 12) = *reinterpret_cast<const float4*>(Wg + k0 + 12);
    __syncthreads();

    const v16bf aF0 = load_frag_f32(&As[0][(wm + 0  + lr) * LDSA_STRIDE], half);
    const v16bf aF1 = load_frag_f32(&As[0][(wm + 16 + lr) * LDSA_STRIDE], half);
    const v16bf bF0 = load_frag_f32(&Bs[0][(wn + 0  + lr) * LDSA_STRIDE], half);
    const v16bf bF1 = load_frag_f32(&Bs[0][(wn + 16 + lr) * LDSA_STRIDE], half);

    acc00 = wmma_bf16(aF0, bF0, acc00);
    acc01 = wmma_bf16(aF0, bF1, acc01);
    acc10 = wmma_bf16(aF1, bF0, acc10);
    acc11 = wmma_bf16(aF1, bF1, acc11);
    __syncthreads();
  }
#endif

  // C/D layout: VGPR v -> row (v + 8*half), col (lane & 15)
  const int n0 = blockN + wn + lr;
  const int n1 = n0 + 16;
  const float bv0 = bias[n0];
  const float bv1 = bias[n1];
#pragma unroll
  for (int v = 0; v < 8; ++v) {
    const int m0 = blockM + wm + v + half * 8;
    const int m1 = m0 + 16;
    C[(size_t)m0 * N + n0] = acc00[v] + bv0;
    C[(size_t)m0 * N + n1] = acc01[v] + bv1;
    C[(size_t)m1 * N + n0] = acc10[v] + bv0;
    C[(size_t)m1 * N + n1] = acc11[v] + bv1;
  }
}

// ---------------------------------------------------------------------------
// Software grid barrier (deterministic per call: counters zeroed by init
// kernel before the recurrence kernel runs). bar[0]=arrive count, bar[1]=sense.
// ---------------------------------------------------------------------------
__global__ void init_barrier(int* bar) {
  if (threadIdx.x < 4) bar[threadIdx.x] = 0;
}

__device__ __forceinline__ void grid_barrier(int* bar, int* lsense) {
  __threadfence();            // make this thread's global stores visible
  __syncthreads();
  if (threadIdx.x == 0) {
    const int want = ++(*lsense);
    if (atomicAdd(&bar[0], 1) == NWG - 1) {
      bar[0] = 0;
      __threadfence();
      atomicExch(&bar[1], want);
    } else {
      while (atomicAdd(&bar[1], 0) < want) __builtin_amdgcn_s_sleep(2);
    }
  }
  __syncthreads();
}

__device__ __forceinline__ float sigf(float x) { return 1.0f / (1.0f + __expf(-x)); }

// ---------------------------------------------------------------------------
// Sequential 2-layer LSTM recurrence. NWG workgroups x 256 threads.
// WG w owns h-slice j in [w*32, w*32+32): it computes all four gate rows
// (i/f/g/o) for that slice, so the cell update is WG-local; h vectors are
// exchanged through global memory with a grid barrier twice per step.
// ---------------------------------------------------------------------------
__global__ void lstm_recurrence(const float* __restrict__ gx0,
                                const float* __restrict__ Whh0,
                                const float* __restrict__ bhh0,
                                const float* __restrict__ Wih1,
                                const float* __restrict__ Whh1,
                                const float* __restrict__ bih1,
                                const float* __restrict__ bhh1,
                                float* __restrict__ hs,
                                float* __restrict__ state,   // h0,c0,h1,c1 (4*512)
                                int* __restrict__ bar) {
  __shared__ float hA[H];
  __shared__ float hB[H];
  __shared__ float gsum[128];
  __shared__ int   lsense_s;

  const int tid   = threadIdx.x;          // 0..255
  const int wg    = blockIdx.x;           // 0..NWG-1
  const int jbase = wg * 32;

  float* __restrict__ h0 = state;
  float* __restrict__ c0 = state + H;
  float* __restrict__ h1 = state + 2 * H;
  float* __restrict__ c1 = state + 3 * H;

  if (tid == 0) lsense_s = 0;
  if (tid < 32) {                          // zero this WG's state slice
    h0[jbase + tid] = 0.0f; c0[jbase + tid] = 0.0f;
    h1[jbase + tid] = 0.0f; c1[jbase + tid] = 0.0f;
  }
  grid_barrier(bar, (int*)&lsense_s);

  const int rl   = tid >> 1;               // 0..127
  const int part = tid & 1;
  const int gate = rl >> 5;                // 0..3
  const int jj   = rl & 31;
  const int j    = jbase + jj;
  const int grow = gate * H + j;           // gate row in the 4H x H weight

  for (int t = 0; t < STEPS; ++t) {
    // ---------------- layer 0: gates = gx0[t] + Whh0 @ h0 + bhh0 ------------
    for (int i = tid; i < H; i += 256) hA[i] = h0[i];
    __syncthreads();
    {
      const float* __restrict__ wr = Whh0 + (size_t)grow * H + part * 256;
      float s = 0.0f;
#pragma unroll 8
      for (int k = 0; k < 256; ++k) s += wr[k] * hA[part * 256 + k];
      s += __shfl_xor(s, 1);
      if (part == 0) gsum[rl] = s;
    }
    __syncthreads();
    if (tid < 32) {
      const int jq = jbase + tid;
      const size_t gb = (size_t)t * G;
      const float iv = gsum[0 * 32 + tid] + gx0[gb + 0 * H + jq] + bhh0[0 * H + jq];
      const float fv = gsum[1 * 32 + tid] + gx0[gb + 1 * H + jq] + bhh0[1 * H + jq];
      const float gv = gsum[2 * 32 + tid] + gx0[gb + 2 * H + jq] + bhh0[2 * H + jq];
      const float ov = gsum[3 * 32 + tid] + gx0[gb + 3 * H + jq] + bhh0[3 * H + jq];
      const float cn = sigf(fv) * c0[jq] + sigf(iv) * tanhf(gv);
      const float hn = sigf(ov) * tanhf(cn);
      c0[jq] = cn;
      h0[jq] = hn;
    }
    grid_barrier(bar, (int*)&lsense_s);

    // ------- layer 1: gates = Wih1 @ h0 + bih1 + Whh1 @ h1 + bhh1 -----------
    for (int i = tid; i < H; i += 256) { hA[i] = h0[i]; hB[i] = h1[i]; }
    __syncthreads();
    {
      const float* __restrict__ wi = Wih1 + (size_t)grow * H + part * 256;
      const float* __restrict__ wh = Whh1 + (size_t)grow * H + part * 256;
      float s = 0.0f;
#pragma unroll 8
      for (int k = 0; k < 256; ++k)
        s += wi[k] * hA[part * 256 + k] + wh[k] * hB[part * 256 + k];
      s += __shfl_xor(s, 1);
      if (part == 0) gsum[rl] = s;
    }
    __syncthreads();
    if (tid < 32) {
      const int jq = jbase + tid;
      const float iv = gsum[0 * 32 + tid] + bih1[0 * H + jq] + bhh1[0 * H + jq];
      const float fv = gsum[1 * 32 + tid] + bih1[1 * H + jq] + bhh1[1 * H + jq];
      const float gv = gsum[2 * 32 + tid] + bih1[2 * H + jq] + bhh1[2 * H + jq];
      const float ov = gsum[3 * 32 + tid] + bih1[3 * H + jq] + bhh1[3 * H + jq];
      const float cn = sigf(fv) * c1[jq] + sigf(iv) * tanhf(gv);
      const float hn = sigf(ov) * tanhf(cn);
      c1[jq] = cn;
      h1[jq] = hn;
      hs[(size_t)t * H + jq] = hn;
    }
    grid_barrier(bar, (int*)&lsense_s);
  }
}

// ---------------------------------------------------------------------------
// outputs = hs @ W_out^T + b_out   (1024 x 512 x 11), plus hn/cn tail copy.
// d_out layout: outputs (1024*11) | hn (h0,h1 = 1024) | cn (c0,c1 = 1024)
// ---------------------------------------------------------------------------
__global__ void output_proj(const float* __restrict__ hs,
                            const float* __restrict__ W_out,
                            const float* __restrict__ b_out,
                            const float* __restrict__ state,
                            float* __restrict__ out) {
  const int idx = blockIdx.x * blockDim.x + threadIdx.x;   // 0..11263
  if (idx < STEPS * OUT_N) {
    const int t = idx / OUT_N;
    const int o = idx - t * OUT_N;
    const float* __restrict__ h = hs + (size_t)t * H;
    const float* __restrict__ w = W_out + (size_t)o * H;
    float s = b_out[o];
#pragma unroll 8
    for (int k = 0; k < H; ++k) s += h[k] * w[k];
    out[idx] = s;
  }
  if (idx < 2 * H) {
    const int which = idx >> 9;          // 0 -> layer0, 1 -> layer1
    const int jq    = idx & (H - 1);
    out[STEPS * OUT_N + idx]         = state[which * 2 * H + jq];       // h0,h1
    out[STEPS * OUT_N + 2 * H + idx] = state[H + which * 2 * H + jq];   // c0,c1
  }
}

// ---------------------------------------------------------------------------
// Launcher
// ---------------------------------------------------------------------------
extern "C" void kernel_launch(void* const* d_in, const int* in_sizes, int n_in,
                              void* d_out, int out_size, void* d_ws, size_t ws_size,
                              hipStream_t stream) {
  (void)in_sizes; (void)n_in; (void)out_size; (void)ws_size;

  const float* inputs = (const float*)d_in[0];
  const float* W_in   = (const float*)d_in[1];
  const float* b_in   = (const float*)d_in[2];
  const float* Wih0   = (const float*)d_in[3];
  const float* Whh0   = (const float*)d_in[4];
  const float* bih0   = (const float*)d_in[5];
  const float* bhh0   = (const float*)d_in[6];
  const float* Wih1   = (const float*)d_in[7];
  const float* Whh1   = (const float*)d_in[8];
  const float* bih1   = (const float*)d_in[9];
  const float* bhh1   = (const float*)d_in[10];
  const float* W_out  = (const float*)d_in[11];
  const float* b_out  = (const float*)d_in[12];

  float* ws     = (float*)d_ws;
  float* x_proj = ws;                                  // 1024*512
  float* gx0    = x_proj + (size_t)STEPS * H;          // 1024*2048
  float* hs     = gx0 + (size_t)STEPS * G;             // 1024*512
  float* state  = hs + (size_t)STEPS * H;              // 4*512
  int*   bar    = (int*)(state + 4 * H);               // barrier counters

  init_barrier<<<1, 32, 0, stream>>>(bar);

  // x_proj = inputs @ W_in^T + b_in    (M=1024, N=512, K=19296 = 603*32)
  wmma_gemm_tiled<<<dim3(H / BN, STEPS / BM), 256, 0, stream>>>(
      inputs, W_in, b_in, x_proj, STEPS, H, IN_DIM);

  // gx0 = x_proj @ Wih0^T + bih0       (M=1024, N=2048, K=512)
  wmma_gemm_tiled<<<dim3(G / BN, STEPS / BM), 256, 0, stream>>>(
      x_proj, Wih0, bih0, gx0, STEPS, G, H);

  lstm_recurrence<<<NWG, 256, 0, stream>>>(
      gx0, Whh0, bhh0, Wih1, Whh1, bih1, bhh1, hs, state, bar);

  output_proj<<<(STEPS * OUT_N + 255) / 256, 256, 0, stream>>>(
      hs, W_out, b_out, state, (float*)d_out);
}
